// RNN_21345987461601
// MI455X (gfx1250) — matrix operand order repaired and checked by reading, not compiled, not run on hardware
//
#include <hip/hip_runtime.h>

typedef __attribute__((ext_vector_type(16))) _Float16 v16h;
typedef __attribute__((ext_vector_type(8)))  float    v8f;
typedef __attribute__((ext_vector_type(4)))  float    v4f;

#define RNN_B   2048
#define RNN_T   512
#define RNN_IN  76
#define RNN_H   10
#define RNN_O   15

// D = A * B + C with A = weight matrix (M = output j, K = input k),
//                    B = activation^T  (K = input k, N = batch lane).
//
// A-layout (16x32 f16): lanes 0-15: M=lane, elem e -> K=e (K=0..15)
//                       lanes 16-31: M=lane-16, elem e -> K=16+e (K=16..31)
// B-layout (32x16 f16): lanes 0-15: K=0..15 (elem e -> K=e),  N=lane
//                       lanes 16-31: K=16..31 (elem e -> 16+e), N=lane-16
// C/D-layout: lane n (0-15): (m=r, n) ; lane 16+n: (m=8+r, n)

// ---- loop-invariant weight A-fragment (branchless clamped loads) -----------
__device__ __forceinline__ v16h make_afrag_w(const float* __restrict__ W,
                                             int nOut, int kDim,
                                             int row, int half, int k0) {
    v16h a;
#pragma unroll
    for (int e = 0; e < 16; ++e) {
        int k = k0 + 16 * half + e;
        float v = W[((row < nOut) ? row : 0) * kDim + ((k < kDim) ? k : 0)];
        a[e] = (_Float16)((row < nOut && k < kDim) ? v : 0.0f);
    }
    return a;
}

// ---- x B-fragment: lane reads 16 contiguous floats of its batch row --------
template <int CHUNK>
__device__ __forceinline__ v16h make_bfrag_x(const float* __restrict__ xrow, int half) {
    v16h b;
    if (CHUNK < 2) {                       // cols CHUNK*32 + 16*half .. +15, all < 76
        const float* p = xrow + CHUNK * 32 + 16 * half;
        v4f q0 = *(const v4f*)(p);
        v4f q1 = *(const v4f*)(p + 4);
        v4f q2 = *(const v4f*)(p + 8);
        v4f q3 = *(const v4f*)(p + 12);
#pragma unroll
        for (int e = 0; e < 4; ++e) {
            b[e]      = (_Float16)q0[e];
            b[e + 4]  = (_Float16)q1[e];
            b[e + 8]  = (_Float16)q2[e];
            b[e + 12] = (_Float16)q3[e];
        }
    } else {                               // cols 64..75 valid only for half==0 lanes
        const float* p = xrow + 64;        // half==1 lanes reread same lines (cache hit)
        v4f q0 = *(const v4f*)(p);
        v4f q1 = *(const v4f*)(p + 4);
        v4f q2 = *(const v4f*)(p + 8);
        const bool z = (half != 0);
#pragma unroll
        for (int e = 0; e < 4; ++e) {
            b[e]      = (_Float16)(z ? 0.0f : q0[e]);
            b[e + 4]  = (_Float16)(z ? 0.0f : q1[e]);
            b[e + 8]  = (_Float16)(z ? 0.0f : q2[e]);
            b[e + 12] = (_Float16)0.0f;    // cols 76..79 / 92..95: padded zero
        }
    }
    return b;
}

// ---- D -> B transpose via half-wave swap (ds_swizzle SWAPX16) --------------
__device__ __forceinline__ float swap16(float v) {
    return __builtin_bit_cast(float,
        __builtin_amdgcn_ds_swizzle(__builtin_bit_cast(int, v), 0x401f));
}

__device__ __forceinline__ v16h make_bfrag_h(const v8f& c, int half) {
    v16h b;
#pragma unroll
    for (int r = 0; r < 8; ++r) {
        float own = c[r];        // h[batch=n][j = r + 8*half]
        float oth = swap16(own); // partner half's j values
        // lanes 0-15: K=r from own regs, K=8+r from lane n+16 (j>=10 rows are 0)
        b[r]     = (_Float16)(half ? 0.0f : own);
        b[r + 8] = (_Float16)(half ? 0.0f : oth);
    }
    return b;   // lanes 16-31 (K=16..31 >= hidden) are all zero
}

// ---- bias accumulator seed: C[m][n] = bias[m], m = r + 8*half --------------
__device__ __forceinline__ v8f make_bias_c(const float* __restrict__ b1,
                                           const float* __restrict__ b2,
                                           int nOut, int half) {
    v8f c;
#pragma unroll
    for (int r = 0; r < 8; ++r) {
        int m  = r + 8 * half;
        int mi = (m < nOut) ? m : 0;
        float v = b1[mi] + (b2 ? b2[mi] : 0.0f);
        c[r] = (m < nOut) ? v : 0.0f;
    }
    return c;
}

__device__ __forceinline__ float fast_tanh(float v) {
#if __has_builtin(__builtin_amdgcn_tanhf)
    return __builtin_amdgcn_tanhf(v);          // v_tanh_f32 (gfx1250 TRANS op)
#else
    float xx = fminf(fmaxf(v, -9.0f), 9.0f);
    float e2 = __builtin_amdgcn_exp2f(2.8853900818f * xx);   // e^(2x)
    return (e2 - 1.0f) * __builtin_amdgcn_rcpf(e2 + 1.0f);
#endif
}

__device__ __forceinline__ void tanh8(v8f& c) {
#pragma unroll
    for (int r = 0; r < 8; ++r) c[r] = fast_tanh(c[r]);
}

#define WMMA(A, Bm, C) \
    __builtin_amdgcn_wmma_f32_16x16x32_f16(false, (A), false, (Bm), (short)0, (C), false, false)

__global__ void __launch_bounds__(32)
rnn_fused_kernel(const float* __restrict__ x,
                 const float* __restrict__ Wih0, const float* __restrict__ Whh0,
                 const float* __restrict__ bih0, const float* __restrict__ bhh0,
                 const float* __restrict__ Wih1, const float* __restrict__ Whh1,
                 const float* __restrict__ bih1, const float* __restrict__ bhh1,
                 const float* __restrict__ Wih2, const float* __restrict__ Whh2,
                 const float* __restrict__ bih2, const float* __restrict__ bhh2,
                 const float* __restrict__ Wfc,  const float* __restrict__ bfc,
                 float* __restrict__ out) {
    const int lane = threadIdx.x;       // 0..31 (wave32)
    const int half = lane >> 4;
    const int n    = lane & 15;         // batch column within tile
    const int b0   = blockIdx.x * 16;

    // Loop-invariant weight A-fragments.
    const v16h a_x0  = make_afrag_w(Wih0, RNN_H, RNN_IN, n, half, 0);
    const v16h a_x1  = make_afrag_w(Wih0, RNN_H, RNN_IN, n, half, 32);
    const v16h a_x2  = make_afrag_w(Wih0, RNN_H, RNN_IN, n, half, 64);
    const v16h a_hh0 = make_afrag_w(Whh0, RNN_H, RNN_H, n, half, 0);
    const v16h a_ih1 = make_afrag_w(Wih1, RNN_H, RNN_H, n, half, 0);
    const v16h a_hh1 = make_afrag_w(Whh1, RNN_H, RNN_H, n, half, 0);
    const v16h a_ih2 = make_afrag_w(Wih2, RNN_H, RNN_H, n, half, 0);
    const v16h a_hh2 = make_afrag_w(Whh2, RNN_H, RNN_H, n, half, 0);
    const v16h a_fc  = make_afrag_w(Wfc,  RNN_O, RNN_H, n, half, 0);

    // Loop-invariant bias accumulator seeds.
    const v8f cb0 = make_bias_c(bih0, bhh0, RNN_H, half);
    const v8f cb1 = make_bias_c(bih1, bhh1, RNN_H, half);
    const v8f cb2 = make_bias_c(bih2, bhh2, RNN_H, half);
    const v8f cbF = make_bias_c(bfc,  nullptr, RNN_O, half);

    // Hidden states as B-fragments; zero start.
    v16h bh0 = {};
    v16h bh1 = {};
    v16h bh2 = {};

    const float* xbase = x + (size_t)(b0 + n) * RNN_T * RNN_IN;

    // Double-buffered x fragments.
    v16h bx0 = make_bfrag_x<0>(xbase, half);
    v16h bx1 = make_bfrag_x<1>(xbase, half);
    v16h bx2 = make_bfrag_x<2>(xbase, half);

#pragma unroll 1
    for (int t = 0; t < RNN_T; ++t) {
        // Issue next step's loads early; they overlap the WMMA/tanh chain.
        const float* xn = xbase + (size_t)((t + 1 < RNN_T) ? t + 1 : t) * RNN_IN;
        v16h nx0 = make_bfrag_x<0>(xn, half);
        v16h nx1 = make_bfrag_x<1>(xn, half);
        v16h nx2 = make_bfrag_x<2>(xn, half);

        // ---- layer 0 ----
        v8f c0 = cb0;
        c0 = WMMA(a_x0,  bx0, c0);
        c0 = WMMA(a_x1,  bx1, c0);
        c0 = WMMA(a_x2,  bx2, c0);
        c0 = WMMA(a_hh0, bh0, c0);
        tanh8(c0);
        bh0 = make_bfrag_h(c0, half);   // n0: layer-1 input and next-step h0

        // ---- layer 1 ----
        v8f c1 = cb1;
        c1 = WMMA(a_ih1, bh0, c1);
        c1 = WMMA(a_hh1, bh1, c1);      // bh1 still holds previous-step h1
        tanh8(c1);
        bh1 = make_bfrag_h(c1, half);

        // ---- layer 2 ----
        v8f c2 = cb2;
        c2 = WMMA(a_ih2, bh1, c2);
        c2 = WMMA(a_hh2, bh2, c2);
        tanh8(c2);
        bh2 = make_bfrag_h(c2, half);

        bx0 = nx0; bx1 = nx1; bx2 = nx2;
    }

    // ---- final FC: out[b0+n][m] = (h2 Wfc^T + bfc), m = r + 8*half ----
    v8f cF = cbF;
    cF = WMMA(a_fc, bh2, cF);
#pragma unroll
    for (int r = 0; r < 8; ++r) {
        int m = r + 8 * half;
        if (m < RNN_O)
            out[(size_t)(b0 + n) * RNN_O + m] = cF[r];
    }
}

extern "C" void kernel_launch(void* const* d_in, const int* in_sizes, int n_in,
                              void* d_out, int out_size, void* d_ws, size_t ws_size,
                              hipStream_t stream) {
    (void)in_sizes; (void)n_in; (void)d_ws; (void)ws_size; (void)out_size;
    const float* x    = (const float*)d_in[0];
    const float* Wih0 = (const float*)d_in[1];
    const float* Whh0 = (const float*)d_in[2];
    const float* bih0 = (const float*)d_in[3];
    const float* bhh0 = (const float*)d_in[4];
    const float* Wih1 = (const float*)d_in[5];
    const float* Whh1 = (const float*)d_in[6];
    const float* bih1 = (const float*)d_in[7];
    const float* bhh1 = (const float*)d_in[8];
    const float* Wih2 = (const float*)d_in[9];
    const float* Whh2 = (const float*)d_in[10];
    const float* bih2 = (const float*)d_in[11];
    const float* bhh2 = (const float*)d_in[12];
    const float* Wfc  = (const float*)d_in[13];
    const float* bfc  = (const float*)d_in[14];
    float* out = (float*)d_out;

    dim3 grid(RNN_B / 16);   // 128 independent batch tiles, one wave each
    dim3 block(32);
    rnn_fused_kernel<<<grid, block, 0, stream>>>(
        x, Wih0, Whh0, bih0, bhh0, Wih1, Whh1, bih1, bhh1,
        Wih2, Whh2, bih2, bhh2, Wfc, bfc, out);
}
